// MultiHeadAttentionBlock_50835232915754
// MI455X (gfx1250) — matrix operand order repaired and checked by reading, not compiled, not run on hardware
//
#include <hip/hip_runtime.h>

#define BB 2
#define LL 2048
#define DD 1024
#define HH 16
#define DKK 64
#define MM (BB * LL) /* 4096 tokens */

typedef __attribute__((ext_vector_type(16))) __bf16 v16bf;
typedef __attribute__((ext_vector_type(8)))  __bf16 v8bf;
typedef __attribute__((ext_vector_type(4)))  __bf16 v4bf;
typedef __attribute__((ext_vector_type(8)))  float  v8f;

// ---------------------------------------------------------------------------
// helpers
// ---------------------------------------------------------------------------
static __device__ __forceinline__ v8f wmma_bf16(v16bf a, v16bf b, v8f c) {
  // D = A(16x32) * B(32x16) + C, f32 accumulate
  return __builtin_amdgcn_wmma_f32_16x16x32_bf16(false, a, false, b, (short)0, c,
                                                 false, false);
}

static __device__ __forceinline__ v16bf concat8(v8bf lo, v8bf hi) {
  return __builtin_shufflevector(lo, hi, 0, 1, 2, 3, 4, 5, 6, 7,
                                 8, 9, 10, 11, 12, 13, 14, 15);
}

// A-matrix fragment 16x32 (MxK) from row-major bf16 src (leading dim ld).
// ISA layout: lanes 0-15 hold M=lane, K chunks {0..7, 16..23};
//             lanes 16-31 hold M=lane-16, K chunks {8..15, 24..31}.
// Both chunks are 16B contiguous -> two b128 loads.
static __device__ __forceinline__ v16bf load_a_bf16(const __bf16* base, int ld) {
  const int lane = threadIdx.x & 31;
  const __bf16* p = base + (size_t)(lane & 15) * ld + ((lane >> 4) << 3);
  const v8bf lo = *(const v8bf*)p;
  const v8bf hi = *(const v8bf*)(p + 16);
  return concat8(lo, hi);
}

// B-matrix fragment 32x16 (KxN) where B[k][n] = src[n*ld + k] (B = src^T).
// lanes 0-15 hold N=lane with K chunks {0..7, 16..23}; lanes 16-31 hold N=lane-16 with K chunks {8..15, 24..31}.
// Both chunks are 16B contiguous -> two b128 loads.
static __device__ __forceinline__ v16bf load_bt_bf16(const __bf16* base, int ld) {
  const int lane = threadIdx.x & 31;
  const __bf16* p = base + (size_t)(lane & 15) * ld + ((lane >> 4) << 3);
  return concat8(*(const v8bf*)p, *(const v8bf*)(p + 16));
}

// ---------------------------------------------------------------------------
// one-time fp32 -> bf16 conversion (bandwidth bound; each element once)
// ---------------------------------------------------------------------------
__global__ __launch_bounds__(256)
void cvt_bf16_kernel(const float* __restrict__ src, __bf16* __restrict__ dst) {
  const size_t i = ((size_t)blockIdx.x * 256 + threadIdx.x) * 4;
  const float4 f = *(const float4*)(src + i);
  v4bf o;
  o[0] = (__bf16)f.x;
  o[1] = (__bf16)f.y;
  o[2] = (__bf16)f.z;
  o[3] = (__bf16)f.w;
  *(v4bf*)(dst + i) = o;
}

// ---------------------------------------------------------------------------
// Projection GEMM: out = X(4096x1024) * W^T + bias (bf16 in, bf16 out).
// split_mode: 1 -> (B,H,L,DK) layout (Q,K)  2 -> (B,H,DK,L) layout (V)
// One 64x64 output tile per wave; 256 threads = 8 waves per block; 128 blocks.
// ---------------------------------------------------------------------------
__global__ __launch_bounds__(256)
void proj_gemm_kernel(const __bf16* __restrict__ X, const __bf16* __restrict__ W,
                      const float* __restrict__ bias, __bf16* __restrict__ out,
                      int split_mode) {
  const int wid = blockIdx.x * 8 + (threadIdx.x >> 5);
  const int m0 = (wid >> 4) * 64;   // M/64 = 64 tiles
  const int n0 = (wid & 15) * 64;   // N/64 = 16 tiles
  v8f zero = {};
  v8f acc[4][4];
#pragma unroll
  for (int i = 0; i < 4; ++i)
#pragma unroll
    for (int j = 0; j < 4; ++j) acc[i][j] = zero;

  for (int k = 0; k < DD; k += 32) {
    v16bf a[4], b[4];
#pragma unroll
    for (int i = 0; i < 4; ++i)
      a[i] = load_a_bf16(X + (size_t)(m0 + 16 * i) * DD + k, DD);
#pragma unroll
    for (int j = 0; j < 4; ++j)
      b[j] = load_bt_bf16(W + (size_t)(n0 + 16 * j) * DD + k, DD);
#pragma unroll
    for (int i = 0; i < 4; ++i)
#pragma unroll
      for (int j = 0; j < 4; ++j) acc[i][j] = wmma_bf16(a[i], b[j], acc[i][j]);
  }

  const int lane = threadIdx.x & 31;
  const int nin = lane & 15;
  const int mb = (lane >> 4) << 3;
#pragma unroll
  for (int i = 0; i < 4; ++i) {
#pragma unroll
    for (int j = 0; j < 4; ++j) {
      const int ncol = n0 + 16 * j + nin;
      const float bv = bias[ncol];
#pragma unroll
      for (int r = 0; r < 8; ++r) {
        const int mrow = m0 + 16 * i + mb + r;
        const float val = acc[i][j][r] + bv;
        const int bi = mrow >> 11, l = mrow & (LL - 1);
        const int h = ncol >> 6, d = ncol & 63;
        size_t dst;
        if (split_mode == 1)
          dst = (((size_t)(bi * HH + h)) * LL + l) * DKK + d;
        else
          dst = (((size_t)(bi * HH + h)) * DKK + d) * LL + l;
        out[dst] = (__bf16)val;
      }
    }
  }
}

// ---------------------------------------------------------------------------
// Output GEMM: out = A(4096x1024 bf16) * Wo^T + bo, fp32 result.
// ---------------------------------------------------------------------------
__global__ __launch_bounds__(256)
void out_gemm_kernel(const __bf16* __restrict__ X, const __bf16* __restrict__ W,
                     const float* __restrict__ bias, float* __restrict__ out) {
  const int wid = blockIdx.x * 8 + (threadIdx.x >> 5);
  const int m0 = (wid >> 4) * 64;
  const int n0 = (wid & 15) * 64;
  v8f zero = {};
  v8f acc[4][4];
#pragma unroll
  for (int i = 0; i < 4; ++i)
#pragma unroll
    for (int j = 0; j < 4; ++j) acc[i][j] = zero;

  for (int k = 0; k < DD; k += 32) {
    v16bf a[4], b[4];
#pragma unroll
    for (int i = 0; i < 4; ++i)
      a[i] = load_a_bf16(X + (size_t)(m0 + 16 * i) * DD + k, DD);
#pragma unroll
    for (int j = 0; j < 4; ++j)
      b[j] = load_bt_bf16(W + (size_t)(n0 + 16 * j) * DD + k, DD);
#pragma unroll
    for (int i = 0; i < 4; ++i)
#pragma unroll
      for (int j = 0; j < 4; ++j) acc[i][j] = wmma_bf16(a[i], b[j], acc[i][j]);
  }

  const int lane = threadIdx.x & 31;
  const int nin = lane & 15;
  const int mb = (lane >> 4) << 3;
#pragma unroll
  for (int i = 0; i < 4; ++i) {
#pragma unroll
    for (int j = 0; j < 4; ++j) {
      const int ncol = n0 + 16 * j + nin;
      const float bv = bias[ncol];
#pragma unroll
      for (int r = 0; r < 8; ++r) {
        const int mrow = m0 + 16 * i + mb + r;
        out[(size_t)mrow * DD + ncol] = acc[i][j][r] + bv;
      }
    }
  }
}

// ---------------------------------------------------------------------------
// Flash attention. Block = (b, h, 64-query tile); 4 waves x 16 query rows.
// KV tile = 128 keys per iteration (amortizes softmax overhead over 32 WMMAs).
// Q,K in (B,H,L,DK) bf16; V in (B,H,DK,L) bf16; output (B,L,D) bf16.
// ---------------------------------------------------------------------------
__global__ __launch_bounds__(128)
void attn_kernel(const __bf16* __restrict__ Qb, const __bf16* __restrict__ Kb,
                 const __bf16* __restrict__ Vb, const int* __restrict__ mask,
                 __bf16* __restrict__ Ob) {
  __shared__ __align__(32) __bf16 plds[4][16][128];  // per-wave P staging (16 KB)

  const int qt = blockIdx.x & 31;          // L/64 = 32 query tiles
  const int h  = (blockIdx.x >> 5) & 15;
  const int b  = blockIdx.x >> 9;
  const int bh = b * HH + h;
  const int w    = threadIdx.x >> 5;
  const int lane = threadIdx.x & 31;
  const int nin  = lane & 15;
  const int mb   = (lane >> 4) << 3;
  const int q0   = qt * 64 + w * 16;
  const float scale = 0.125f;              // 1/sqrt(64)

  const __bf16* Qp = Qb + ((size_t)bh * LL + q0) * DKK;
  const v16bf qa0 = load_a_bf16(Qp, DKK);
  const v16bf qa1 = load_a_bf16(Qp + 32, DKK);

  v8f zero = {};
  v8f o[4];
#pragma unroll
  for (int jn = 0; jn < 4; ++jn) o[jn] = zero;
  float mrow[8], lrow[8];
#pragma unroll
  for (int r = 0; r < 8; ++r) { mrow[r] = -3.0e38f; lrow[r] = 0.f; }

  for (int jt = 0; jt < LL / 128; ++jt) {
    const int kbase = jt * 128;

    // S = scale * Q K^T  (16x128 tile, masked)
    v8f s[8];
#pragma unroll
    for (int jn = 0; jn < 8; ++jn) {
      const __bf16* Kp = Kb + ((size_t)bh * LL + kbase + jn * 16) * DKK;
      v8f acc = zero;
      acc = wmma_bf16(qa0, load_bt_bf16(Kp, DKK), acc);
      acc = wmma_bf16(qa1, load_bt_bf16(Kp + 32, DKK), acc);
      const int key = kbase + jn * 16 + nin;
      const bool keep = (mask[b * LL + key] != 0);
#pragma unroll
      for (int r = 0; r < 8; ++r) acc[r] = keep ? acc[r] * scale : -1.0e9f;
      s[jn] = acc;
    }

    // online softmax: per-row max / sum via 16-lane butterflies
#pragma unroll
    for (int r = 0; r < 8; ++r) {
      float mx = s[0][r];
#pragma unroll
      for (int jn = 1; jn < 8; ++jn) mx = fmaxf(mx, s[jn][r]);
#pragma unroll
      for (int t = 1; t < 16; t <<= 1) mx = fmaxf(mx, __shfl_xor(mx, t, 32));
      const float mnew = fmaxf(mrow[r], mx);
      const float corr = __expf(mrow[r] - mnew);
      mrow[r] = mnew;
      float rs = 0.f;
#pragma unroll
      for (int jn = 0; jn < 8; ++jn) {
        const float p = __expf(s[jn][r] - mnew);
        s[jn][r] = p;
        rs += p;
      }
#pragma unroll
      for (int t = 1; t < 16; t <<= 1) rs += __shfl_xor(rs, t, 32);
      lrow[r] = lrow[r] * corr + rs;
#pragma unroll
      for (int jn = 0; jn < 4; ++jn) o[jn][r] *= corr;
    }

    // C-layout P -> LDS (row-major bf16) -> A-layout fragments
#pragma unroll
    for (int jn = 0; jn < 8; ++jn)
#pragma unroll
      for (int r = 0; r < 8; ++r)
        plds[w][mb + r][jn * 16 + nin] = (__bf16)s[jn][r];
    __syncthreads();

    v16bf pa[4];
    {
      const __bf16* row = &plds[w][lane & 15][0];
      const int ko = (lane >> 4) << 3;
#pragma unroll
      for (int ks = 0; ks < 4; ++ks)
        pa[ks] = concat8(*(const v8bf*)(row + ks * 32 + ko),
                         *(const v8bf*)(row + ks * 32 + ko + 16));
    }
    __syncthreads();

    // O += P * V   (V transposed layout -> contiguous B-fragment loads)
#pragma unroll
    for (int jn = 0; jn < 4; ++jn) {
      const __bf16* Vp = Vb + ((size_t)bh * DKK + jn * 16) * LL + kbase;
#pragma unroll
      for (int ks = 0; ks < 4; ++ks)
        o[jn] = wmma_bf16(pa[ks], load_bt_bf16(Vp + ks * 32, LL), o[jn]);
    }
  }

  // epilogue: normalize and store (B,L,D) bf16
#pragma unroll
  for (int r = 0; r < 8; ++r) {
    const float inv = 1.0f / lrow[r];
    const int qrow = q0 + mb + r;
#pragma unroll
    for (int jn = 0; jn < 4; ++jn) {
      const int d = jn * 16 + nin;
      Ob[((size_t)b * LL + qrow) * DD + h * DKK + d] = (__bf16)(o[jn][r] * inv);
    }
  }
}

// ---------------------------------------------------------------------------
// launch
// ---------------------------------------------------------------------------
extern "C" void kernel_launch(void* const* d_in, const int* in_sizes, int n_in,
                              void* d_out, int out_size, void* d_ws, size_t ws_size,
                              hipStream_t stream) {
  (void)in_sizes; (void)n_in; (void)out_size; (void)ws_size;

  const float* q    = (const float*)d_in[0];
  const float* k    = (const float*)d_in[1];
  const float* v    = (const float*)d_in[2];
  const int*   mask = (const int*)  d_in[3];
  const float* wq   = (const float*)d_in[4];
  const float* bq   = (const float*)d_in[5];
  const float* wk   = (const float*)d_in[6];
  const float* bk   = (const float*)d_in[7];
  const float* wv   = (const float*)d_in[8];
  const float* bv   = (const float*)d_in[9];
  const float* wo   = (const float*)d_in[10];
  const float* bo   = (const float*)d_in[11];
  float* out = (float*)d_out;

  // workspace layout (bf16 elements); total 32M elems = 64 MB
  const size_t ACT = (size_t)MM * DD;  // 4M elems
  const size_t WSZ = (size_t)DD * DD;  // 1M elems
  __bf16* qbf = (__bf16*)d_ws;
  __bf16* kbf = qbf + ACT;
  __bf16* vbf = kbf + ACT;
  __bf16* wqb = vbf + ACT;
  __bf16* wkb = wqb + WSZ;
  __bf16* wvb = wkb + WSZ;
  __bf16* wob = wvb + WSZ;
  __bf16* Qb  = wob + WSZ;
  __bf16* Kb  = Qb + ACT;
  __bf16* Vb  = Kb + ACT;
  __bf16* Ab  = Vb + ACT;

  // one-time conversions (each element touched once)
  cvt_bf16_kernel<<<(int)(ACT / 1024), 256, 0, stream>>>(q, qbf);
  cvt_bf16_kernel<<<(int)(ACT / 1024), 256, 0, stream>>>(k, kbf);
  cvt_bf16_kernel<<<(int)(ACT / 1024), 256, 0, stream>>>(v, vbf);
  cvt_bf16_kernel<<<(int)(WSZ / 1024), 256, 0, stream>>>(wq, wqb);
  cvt_bf16_kernel<<<(int)(WSZ / 1024), 256, 0, stream>>>(wk, wkb);
  cvt_bf16_kernel<<<(int)(WSZ / 1024), 256, 0, stream>>>(wv, wvb);
  cvt_bf16_kernel<<<(int)(WSZ / 1024), 256, 0, stream>>>(wo, wob);

  proj_gemm_kernel<<<128, 256, 0, stream>>>(qbf, wqb, bq, Qb, 1);
  proj_gemm_kernel<<<128, 256, 0, stream>>>(kbf, wkb, bk, Kb, 1);
  proj_gemm_kernel<<<128, 256, 0, stream>>>(vbf, wvb, bv, Vb, 2);
  attn_kernel<<<BB * HH * (LL / 64), 128, 0, stream>>>(Qb, Kb, Vb, mask, Ab);
  out_gemm_kernel<<<128, 256, 0, stream>>>(Ab, wob, bo, out);
}
